// MultiHeadAttention_54889682043406
// MI455X (gfx1250) — matrix operand order repaired and checked by reading, not compiled, not run on hardware
//
#include <hip/hip_runtime.h>
#include <hip/hip_bf16.h>

// Problem constants (from reference)
#define B_  2
#define S_  4096
#define D_  512
#define H_  8
#define DK_ 64

typedef _Float16 v16h __attribute__((ext_vector_type(16)));
typedef float    v8f  __attribute__((ext_vector_type(8)));
typedef int      v4i  __attribute__((ext_vector_type(4)));

// ---------------------------------------------------------------------------
// CDNA5 async global->LDS copy path (ASYNCcnt-tracked DMA).
// Builtin signature (from compiler diagnostic): param0 = v4i AS(1)* (global),
// param1 = v4i AS(3)* (LDS), then imm offset, imm cpol.
// Guarded: falls back to plain vector copies if the toolchain lacks builtins.
// ---------------------------------------------------------------------------
#if __has_builtin(__builtin_amdgcn_global_load_async_to_lds_b128) && \
    __has_builtin(__builtin_amdgcn_s_wait_asynccnt)
#define HAVE_ASYNC_LDS 1
#else
#define HAVE_ASYNC_LDS 0
#endif

typedef __attribute__((address_space(1))) v4i as1_v4i;   // "__device__" ptr
typedef __attribute__((address_space(3))) v4i as3_v4i;   // "__shared__" ptr

__device__ inline void wait_async_copies() {
#if HAVE_ASYNC_LDS
  __builtin_amdgcn_s_wait_asynccnt(0);         // s_wait_asynccnt 0
#endif
}

// Copy one 16-byte chunk global->LDS (async DMA if available).
__device__ inline void copy16(_Float16* lds_dst, const _Float16* gsrc) {
#if HAVE_ASYNC_LDS
  // global generic pointer bits == AS1 bits; generic LDS pointer low 32 bits
  // are the wave-relative LDS offset (ISA 10.2 aperture rules).
  __builtin_amdgcn_global_load_async_to_lds_b128(
      (as1_v4i*)(size_t)gsrc,
      (as3_v4i*)(unsigned)(size_t)lds_dst,
      0, 0);
#else
  *(uint4*)lds_dst = *(const uint4*)gsrc;
#endif
}

// Stage a 64x64 f16 tile (row-major, global stride DK_) into LDS with stride 72.
// 512 16B-chunks across 128 threads.
__device__ inline void stage_tile64(_Float16* __restrict__ dst,
                                    const _Float16* __restrict__ src, int tid) {
#pragma unroll
  for (int i = 0; i < 4; ++i) {
    const int chunk = tid + i * 128;           // 0..511
    const int r = chunk >> 3, c = (chunk & 7) << 3;
    copy16(dst + r * 72 + c, src + (size_t)r * DK_ + c);
  }
}

// ---------------------------------------------------------------------------
// WMMA fragment gathers (wave32 V_WMMA_F32_16X16X32_F16 layouts)
// A 16x32: lane L row M=L%16; half h -> K = (h<8?h:h+8) + 8*(L>=16)
// B 32x16: lane L col N=L%16; half h -> K = (h<8?h:h+8) + 8*(L>=16)
// C 16x16 f32: lane L col N=L%16; vgpr r row M = r + 8*(L>=16)
// ---------------------------------------------------------------------------
__device__ inline v16h load_frag_a(const _Float16* __restrict__ t, int stride, int lane) {
  const int row  = lane & 15;
  const int koff = (lane >> 4) << 3;
  const _Float16* p = t + row * stride;
  v16h a;
#pragma unroll
  for (int j = 0; j < 8; ++j) {
    const int k = ((j < 4) ? (2 * j) : (2 * j + 8)) + koff;
    a[2 * j]     = p[k];
    a[2 * j + 1] = p[k + 1];
  }
  return a;
}

__device__ inline v16h load_frag_b(const _Float16* __restrict__ t, int stride, int lane) {
  const int row  = lane & 15;
  const int koff = (lane >> 4) << 3;
  const _Float16* p = t + row * stride;
  v16h b;
#pragma unroll
  for (int j = 0; j < 8; ++j) {
    const int k = ((j < 4) ? (2 * j) : (2 * j + 8)) + koff;
    b[2 * j]     = p[k];
    b[2 * j + 1] = p[k + 1];
  }
  return b;
}

__device__ inline v8f wmma16(v16h a, v16h b, v8f c) {
  return __builtin_amdgcn_wmma_f32_16x16x32_f16(false, a, false, b, (short)0, c, false, false);
}

// ---------------------------------------------------------------------------
// Kernel 1: fused QKV projection (f32 in, f16 [B,H,S,DK] out).
// 64x64 tile/block, 4 waves each a 32x32 quadrant, K step 32.
// ---------------------------------------------------------------------------
__global__ __launch_bounds__(128) void qkv_proj_kernel(
    const float* __restrict__ Q, const float* __restrict__ K, const float* __restrict__ V,
    const float* __restrict__ Wq, const float* __restrict__ bq,
    const float* __restrict__ Wk, const float* __restrict__ bk,
    const float* __restrict__ Wv, const float* __restrict__ bv,
    _Float16* __restrict__ qh, _Float16* __restrict__ kh, _Float16* __restrict__ vh)
{
  const int which = blockIdx.z;
  const float* X    = (which == 0) ? Q  : ((which == 1) ? K  : V);
  const float* W    = (which == 0) ? Wq : ((which == 1) ? Wk : Wv);
  const float* bias = (which == 0) ? bq : ((which == 1) ? bk : bv);
  _Float16*    out  = (which == 0) ? qh : ((which == 1) ? kh : vh);

  const int m0 = blockIdx.x * 64;
  const int n0 = blockIdx.y * 64;
  const int tid = threadIdx.x, lane = tid & 31, wave = tid >> 5;
  const int wr = (wave >> 1) * 32, wc = (wave & 1) * 32;

  __shared__ _Float16 At[64 * 34];
  __shared__ _Float16 Bt[64 * 34];

  v8f acc[2][2] = {};

  for (int k0 = 0; k0 < D_; k0 += 32) {
#pragma unroll
    for (int i = 0; i < 16; ++i) {
      int idx = tid + i * 128;
      int r = idx >> 5, c = idx & 31;
      At[r * 34 + c] = (_Float16)X[(size_t)(m0 + r) * D_ + k0 + c];
      Bt[r * 34 + c] = (_Float16)W[(size_t)(n0 + r) * D_ + k0 + c];
    }
    __syncthreads();
    v16h a0 = load_frag_a(At + (wr + 0)  * 34, 34, lane);
    v16h a1 = load_frag_a(At + (wr + 16) * 34, 34, lane);
    v16h b0 = load_frag_b(Bt + (wc + 0)  * 34, 34, lane);
    v16h b1 = load_frag_b(Bt + (wc + 16) * 34, 34, lane);
    acc[0][0] = wmma16(a0, b0, acc[0][0]);
    acc[0][1] = wmma16(a0, b1, acc[0][1]);
    acc[1][0] = wmma16(a1, b0, acc[1][0]);
    acc[1][1] = wmma16(a1, b1, acc[1][1]);
    __syncthreads();
  }

#pragma unroll
  for (int tr = 0; tr < 2; ++tr)
#pragma unroll
    for (int tc = 0; tc < 2; ++tc) {
      const int nn = n0 + wc + tc * 16 + (lane & 15);
      const int hh = nn >> 6, dk = nn & 63;
      const float bv_ = bias[nn];
#pragma unroll
      for (int r = 0; r < 8; ++r) {
        const int mm = m0 + wr + tr * 16 + ((lane >> 4) << 3) + r;
        const int bb = mm >> 12, ss = mm & (S_ - 1);
        out[(((size_t)bb * H_ + hh) * S_ + ss) * DK_ + dk] =
            (_Float16)(acc[tr][tc][r] + bv_);
      }
    }
}

// ---------------------------------------------------------------------------
// Kernel 2: fused attention.  Per (b,h, 64-row strip):
//  pass 1: stream K tiles (async double-buffered), WMMA scores, online
//          softmax stats only (no score materialization).
//  pass 2: re-stream K (recompute scores: cheaper than 2 extra GB of HBM
//          traffic), normalize with final stats, write attn ONCE (1 GB,
//          mandatory output), convert p->f16 and accumulate context = p @ v.
// ---------------------------------------------------------------------------
__global__ __launch_bounds__(128) void attention_kernel(
    const _Float16* __restrict__ qh, const _Float16* __restrict__ kh,
    const _Float16* __restrict__ vh, const unsigned char* __restrict__ mask,
    float* __restrict__ attn, _Float16* __restrict__ ctx)
{
  const int bh = blockIdx.y;
  const int b = bh >> 3, h = bh & 7;
  const int m0 = blockIdx.x * 64;
  const int tid = threadIdx.x, lane = tid & 31, wave = tid >> 5;
  const int wr = (wave >> 1) * 32, wc = (wave & 1) * 32;

  const _Float16* qb = qh + ((size_t)bh * S_ + m0) * DK_;
  const _Float16* kb = kh + (size_t)bh * S_ * DK_;
  const _Float16* vb = vh + (size_t)bh * S_ * DK_;
  float* ab = attn + (size_t)bh * S_ * S_;
  const unsigned char* mb = mask + (size_t)b * S_ * S_ + (size_t)m0 * S_;

  __shared__ _Float16 Qt[64 * 72];          // resident q tile
  __shared__ _Float16 Kt[2][64 * 72];       // double-buffered k tiles
  __shared__ _Float16 Vt[64 * 72];          // v tile, dk-major (transposed)
  __shared__ float    Sc[64 * 72];          // pass-1 scores; pass-2 aliased as Pt
  __shared__ float    rmS[64], rliS[64];
  _Float16* Pt = (_Float16*)Sc;

  stage_tile64(Qt, qb, tid);
  stage_tile64(Kt[0], kb, tid);
  float rm = -3.0e38f, rl = 0.0f;

  // ---------------- pass 1: softmax stats ----------------
  for (int it = 0; it < S_ / 64; ++it) {
    wait_async_copies();
    __syncthreads();
    if (it + 1 < S_ / 64)
      stage_tile64(Kt[(it + 1) & 1], kb + (size_t)(it + 1) * 64 * DK_, tid);
    const _Float16* kt = Kt[it & 1];
    const int n0 = it * 64;

    v8f sacc[2][2] = {};
#pragma unroll
    for (int ks = 0; ks < 2; ++ks) {
      v16h a0 = load_frag_a(Qt + (wr + 0)  * 72 + ks * 32, 72, lane);
      v16h a1 = load_frag_a(Qt + (wr + 16) * 72 + ks * 32, 72, lane);
      v16h b0 = load_frag_b(kt + (wc + 0)  * 72 + ks * 32, 72, lane);
      v16h b1 = load_frag_b(kt + (wc + 16) * 72 + ks * 32, 72, lane);
      sacc[0][0] = wmma16(a0, b0, sacc[0][0]);
      sacc[0][1] = wmma16(a0, b1, sacc[0][1]);
      sacc[1][0] = wmma16(a1, b0, sacc[1][0]);
      sacc[1][1] = wmma16(a1, b1, sacc[1][1]);
    }

#pragma unroll
    for (int tr = 0; tr < 2; ++tr)
#pragma unroll
      for (int tc = 0; tc < 2; ++tc) {
        const int nl = wc + tc * 16 + (lane & 15);
#pragma unroll
        for (int r = 0; r < 8; ++r) {
          const int ml = wr + tr * 16 + ((lane >> 4) << 3) + r;
          float s = sacc[tr][tc][r] * 0.125f;
          if (mb[(size_t)ml * S_ + n0 + nl]) s = -1.0e9f;
          Sc[ml * 72 + nl] = s;
        }
      }
    __syncthreads();

    if (tid < 64) {                          // one row per thread
      float tmax = -3.0e38f;
#pragma unroll 8
      for (int j = 0; j < 64; ++j) tmax = fmaxf(tmax, Sc[tid * 72 + j]);
      const float mn = fmaxf(rm, tmax);
      float s = 0.0f;
#pragma unroll 8
      for (int j = 0; j < 64; ++j) s += __expf(Sc[tid * 72 + j] - mn);
      rl = rl * __expf(rm - mn) + s;
      rm = mn;
    }
    __syncthreads();
  }

  if (tid < 64) { rmS[tid] = rm; rliS[tid] = 1.0f / rl; }
  stage_tile64(Kt[0], kb, tid);              // restart K pipeline

  // ---------------- pass 2: attn write + context ----------------
  v8f cacc[2][2] = {};
  for (int it = 0; it < S_ / 64; ++it) {
    wait_async_copies();
    __syncthreads();
    if (it + 1 < S_ / 64)
      stage_tile64(Kt[(it + 1) & 1], kb + (size_t)(it + 1) * 64 * DK_, tid);
    const _Float16* kt = Kt[it & 1];
    const int n0 = it * 64;

    if (it + 1 < S_ / 64)                    // global_prefetch_b8 of next v tile
      __builtin_prefetch(&vb[(size_t)(n0 + 64) * DK_ + tid], 0, 1);

    // stage v tile transposed (dk-major) for contiguous B-fragments
#pragma unroll
    for (int i = 0; i < 32; ++i) {
      int idx = tid + i * 128;               // 0..4095
      int kk = idx >> 6, dk = idx & 63;
      Vt[dk * 72 + kk] = vb[(size_t)(n0 + kk) * DK_ + dk];
    }

    // recompute scores
    v8f sacc[2][2] = {};
#pragma unroll
    for (int ks = 0; ks < 2; ++ks) {
      v16h a0 = load_frag_a(Qt + (wr + 0)  * 72 + ks * 32, 72, lane);
      v16h a1 = load_frag_a(Qt + (wr + 16) * 72 + ks * 32, 72, lane);
      v16h b0 = load_frag_b(kt + (wc + 0)  * 72 + ks * 32, 72, lane);
      v16h b1 = load_frag_b(kt + (wc + 16) * 72 + ks * 32, 72, lane);
      sacc[0][0] = wmma16(a0, b0, sacc[0][0]);
      sacc[0][1] = wmma16(a0, b1, sacc[0][1]);
      sacc[1][0] = wmma16(a1, b0, sacc[1][0]);
      sacc[1][1] = wmma16(a1, b1, sacc[1][1]);
    }

    // normalize, emit final attn probabilities, stage P tile (f16)
#pragma unroll
    for (int tr = 0; tr < 2; ++tr)
#pragma unroll
      for (int tc = 0; tc < 2; ++tc) {
        const int nl = wc + tc * 16 + (lane & 15);
#pragma unroll
        for (int r = 0; r < 8; ++r) {
          const int ml = wr + tr * 16 + ((lane >> 4) << 3) + r;
          float s = sacc[tr][tc][r] * 0.125f;
          if (mb[(size_t)ml * S_ + n0 + nl]) s = -1.0e9f;
          const float p = __expf(s - rmS[ml]) * rliS[ml];
          ab[(size_t)(m0 + ml) * S_ + n0 + nl] = p;
          Pt[ml * 72 + nl] = (_Float16)p;
        }
      }
    __syncthreads();

    // context += P @ V   (K = 64 -> two 32-wide steps)
#pragma unroll
    for (int ks = 0; ks < 2; ++ks) {
      v16h a0 = load_frag_a(Pt + (wr + 0)  * 72 + ks * 32, 72, lane);
      v16h a1 = load_frag_a(Pt + (wr + 16) * 72 + ks * 32, 72, lane);
      v16h b0 = load_frag_b(Vt + (wc + 0)  * 72 + ks * 32, 72, lane);
      v16h b1 = load_frag_b(Vt + (wc + 16) * 72 + ks * 32, 72, lane);
      cacc[0][0] = wmma16(a0, b0, cacc[0][0]);
      cacc[0][1] = wmma16(a0, b1, cacc[0][1]);
      cacc[1][0] = wmma16(a1, b0, cacc[1][0]);
      cacc[1][1] = wmma16(a1, b1, cacc[1][1]);
    }
  }

#pragma unroll
  for (int tr = 0; tr < 2; ++tr)
#pragma unroll
    for (int tc = 0; tc < 2; ++tc) {
      const int dk = wc + tc * 16 + (lane & 15);
#pragma unroll
      for (int r = 0; r < 8; ++r) {
        const int ml = wr + tr * 16 + ((lane >> 4) << 3) + r;
        ctx[((size_t)b * S_ + m0 + ml) * D_ + h * DK_ + dk] = (_Float16)cacc[tr][tc][r];
      }
    }
}

// ---------------------------------------------------------------------------
// Kernel 3: x = ctx @ Wo^T + bo + Q (residual).  ctx (f16) staged via async
// LDS DMA (stride 40 halfs = 80B rows keeps 16B alignment, conflict-free).
// ---------------------------------------------------------------------------
__global__ __launch_bounds__(128) void outproj_kernel(
    const _Float16* __restrict__ ctx, const float* __restrict__ Wo,
    const float* __restrict__ bo, const float* __restrict__ Qin,
    float* __restrict__ xres)
{
  const int m0 = blockIdx.x * 64;
  const int n0 = blockIdx.y * 64;
  const int tid = threadIdx.x, lane = tid & 31, wave = tid >> 5;
  const int wr = (wave >> 1) * 32, wc = (wave & 1) * 32;

  __shared__ _Float16 At[64 * 40];
  __shared__ _Float16 Bt[64 * 34];

  v8f acc[2][2] = {};

  for (int k0 = 0; k0 < D_; k0 += 32) {
#pragma unroll
    for (int i = 0; i < 2; ++i) {           // 256 16B chunks, 64x32 f16 A tile
      const int chunk = tid + i * 128;
      const int r = chunk >> 2, c = (chunk & 3) << 3;
      copy16(At + r * 40 + c, ctx + (size_t)(m0 + r) * D_ + k0 + c);
    }
#pragma unroll
    for (int i = 0; i < 16; ++i) {          // Wo f32 -> f16 conversion
      int idx = tid + i * 128;
      int r = idx >> 5, c = idx & 31;
      Bt[r * 34 + c] = (_Float16)Wo[(size_t)(n0 + r) * D_ + k0 + c];
    }
    wait_async_copies();
    __syncthreads();
    v16h a0 = load_frag_a(At + (wr + 0)  * 40, 40, lane);
    v16h a1 = load_frag_a(At + (wr + 16) * 40, 40, lane);
    v16h b0 = load_frag_b(Bt + (wc + 0)  * 34, 34, lane);
    v16h b1 = load_frag_b(Bt + (wc + 16) * 34, 34, lane);
    acc[0][0] = wmma16(a0, b0, acc[0][0]);
    acc[0][1] = wmma16(a0, b1, acc[0][1]);
    acc[1][0] = wmma16(a1, b0, acc[1][0]);
    acc[1][1] = wmma16(a1, b1, acc[1][1]);
    __syncthreads();
  }

#pragma unroll
  for (int tr = 0; tr < 2; ++tr)
#pragma unroll
    for (int tc = 0; tc < 2; ++tc) {
      const int nn = n0 + wc + tc * 16 + (lane & 15);
      const float bo_ = bo[nn];
#pragma unroll
      for (int r = 0; r < 8; ++r) {
        const int mm = m0 + wr + tr * 16 + ((lane >> 4) << 3) + r;
        xres[(size_t)mm * D_ + nn] =
            acc[tr][tc][r] + bo_ + Qin[(size_t)mm * D_ + nn];
      }
    }
}

// ---------------------------------------------------------------------------
// Kernel 4: LayerNorm per row of 512, one block per row.
// ---------------------------------------------------------------------------
__global__ __launch_bounds__(128) void layernorm_kernel(
    const float* __restrict__ x, const float* __restrict__ gamma,
    const float* __restrict__ beta, float* __restrict__ out)
{
  const int row = blockIdx.x;
  const int tid = threadIdx.x;
  const float* xr = x + (size_t)row * D_;

  float v[4], s = 0.0f, ss = 0.0f;
#pragma unroll
  for (int i = 0; i < 4; ++i) {
    v[i] = xr[tid + i * 128];
    s += v[i];
    ss += v[i] * v[i];
  }
  __shared__ float s1[128], s2[128];
  s1[tid] = s; s2[tid] = ss;
  __syncthreads();
  for (int off = 64; off > 0; off >>= 1) {
    if (tid < off) { s1[tid] += s1[tid + off]; s2[tid] += s2[tid + off]; }
    __syncthreads();
  }
  const float mu   = s1[0] * (1.0f / D_);
  const float var  = s2[0] * (1.0f / D_) - mu * mu;
  const float rstd = rsqrtf(var + 1e-5f);
  float* orow = out + (size_t)row * D_;
#pragma unroll
  for (int i = 0; i < 4; ++i) {
    const int j = tid + i * 128;
    orow[j] = (v[i] - mu) * rstd * gamma[j] + beta[j];
  }
}

// ---------------------------------------------------------------------------
extern "C" void kernel_launch(void* const* d_in, const int* in_sizes, int n_in,
                              void* d_out, int out_size, void* d_ws, size_t ws_size,
                              hipStream_t stream)
{
  const float* Q  = (const float*)d_in[0];
  const float* K  = (const float*)d_in[1];
  const float* V  = (const float*)d_in[2];
  const unsigned char* mask = (const unsigned char*)d_in[3];
  const float* Wq = (const float*)d_in[4];
  const float* bq = (const float*)d_in[5];
  const float* Wk = (const float*)d_in[6];
  const float* bk = (const float*)d_in[7];
  const float* Wv = (const float*)d_in[8];
  const float* bv = (const float*)d_in[9];
  const float* Wo = (const float*)d_in[10];
  const float* bo = (const float*)d_in[11];
  const float* gamma = (const float*)d_in[12];
  const float* beta  = (const float*)d_in[13];

  float* out_normed = (float*)d_out;                       // B*S*D
  float* out_attn   = out_normed + (size_t)B_ * S_ * D_;   // B*H*S*S

  char* ws = (char*)d_ws;
  const size_t QKV_BYTES = (size_t)B_ * H_ * S_ * DK_ * sizeof(_Float16); // 8 MiB
  _Float16* qh   = (_Float16*)(ws + 0);
  _Float16* kh   = (_Float16*)(ws + QKV_BYTES);
  _Float16* vh   = (_Float16*)(ws + 2 * QKV_BYTES);
  _Float16* ctx  = (_Float16*)(ws + 3 * QKV_BYTES);
  float*    xres = (float*)   (ws + 4 * QKV_BYTES);

  dim3 blk(128);

  qkv_proj_kernel<<<dim3(B_ * S_ / 64, D_ / 64, 3), blk, 0, stream>>>(
      Q, K, V, Wq, bq, Wk, bk, Wv, bv, qh, kh, vh);

  attention_kernel<<<dim3(S_ / 64, B_ * H_), blk, 0, stream>>>(
      qh, kh, vh, mask, out_attn, ctx);

  outproj_kernel<<<dim3(B_ * S_ / 64, D_ / 64), blk, 0, stream>>>(
      ctx, Wo, bo, Q, xres);

  layernorm_kernel<<<dim3(B_ * S_), blk, 0, stream>>>(
      xres, gamma, beta, out_normed);
}
